// SparseCNNProjectionNetwork_15650860827379
// MI455X (gfx1250) — matrix operand order repaired and checked by reading, not compiled, not run on hardware
//
#include <hip/hip_runtime.h>
#include <hip/hip_bf16.h>

typedef __attribute__((ext_vector_type(16))) _Float16 v16h;
typedef __attribute__((ext_vector_type(8)))  float    v8f;

#define GRID_B 16
#define GRID_H 512
#define GRID_W 512

// ---------------------------------------------------------------- map / nbr
__global__ __launch_bounds__(256) void init_map_kernel(int* map, int total) {
    for (int i = blockIdx.x * 256 + threadIdx.x; i < total; i += gridDim.x * 256)
        map[i] = -1;
}

__global__ __launch_bounds__(256) void scatter_map_kernel(const int* __restrict__ idx,
                                                          int* __restrict__ map, int n) {
    int i = blockIdx.x * 256 + threadIdx.x;
    if (i >= n) return;
    int b = idx[3 * i], y = idx[3 * i + 1], x = idx[3 * i + 2];
    map[((size_t)b * GRID_H + y) * GRID_W + x] = i;
}

__global__ __launch_bounds__(256) void build_nbr_kernel(const int* __restrict__ idx,
                                                        const int* __restrict__ map,
                                                        int* __restrict__ nbr, int n) {
    int i = blockIdx.x * 256 + threadIdx.x;
    if (i >= n) return;
    int b = idx[3 * i], y = idx[3 * i + 1], x = idx[3 * i + 2];
    const int* mb = map + (size_t)b * GRID_H * GRID_W;
#pragma unroll
    for (int ky = 0; ky < 3; ++ky)
#pragma unroll
        for (int kx = 0; kx < 3; ++kx) {
            int yy = y + ky - 1, xx = x + kx - 1;
            int j = (yy >= 0 && yy < GRID_H && xx >= 0 && xx < GRID_W)
                        ? mb[yy * GRID_W + xx] : -1;
            nbr[(size_t)i * 9 + ky * 3 + kx] = j;
        }
}

// ---------------------------------------------------------------- conv1 (cin=1): scalar, tiny
__global__ __launch_bounds__(256) void conv1_kernel(const float* __restrict__ feats,
                                                    const int* __restrict__ nbr,
                                                    const float* __restrict__ W1,
                                                    float* __restrict__ y, int n) {
    __shared__ float w1s[72];
    if (threadIdx.x < 72) w1s[threadIdx.x] = W1[threadIdx.x];
    __syncthreads();
    int i = blockIdx.x * 256 + threadIdx.x;
    if (i >= n) return;
    float acc[8] = {};
#pragma unroll
    for (int t = 0; t < 9; ++t) {
        int j = nbr[(size_t)i * 9 + t];
        if (j >= 0) {
            float f = feats[j];
#pragma unroll
            for (int co = 0; co < 8; ++co) acc[co] += f * w1s[t * 8 + co];
        }
    }
#pragma unroll
    for (int co = 0; co < 8; ++co) y[(size_t)i * 8 + co] = acc[co];
}

// ---------------------------------------------------------------- conv2/conv3: implicit-GEMM WMMA
// One workgroup = 8 waves = 8 tiles of 16 voxels. A (16 x KPAD f16 patch) and
// B (weights [COUT][KPAD] f16, K-contiguous per column) staged in LDS so each
// lane fragment is two b128 LDS loads. Accumulate in f32 via v_wmma_f32_16x16x32_f16.
template <int CIN, int COUT, int KPAD>
__global__ __launch_bounds__(256) void conv_wmma_kernel(const _Float16* __restrict__ xh,
                                                        const int* __restrict__ nbr,
                                                        const float* __restrict__ Wg,
                                                        float* __restrict__ y, int n) {
    constexpr int K   = 9 * CIN;
    constexpr int PST = KPAD + 8;  // +16B row pad: breaks LDS bank conflicts
    __shared__ __align__(16) _Float16 patch[8 * 16 * PST];
    __shared__ __align__(16) _Float16 wt[COUT * KPAD];

    const int tid = threadIdx.x;

    // zero patch (pad taps / invalid neighbors stay 0)
    {
        uint4 z = {0u, 0u, 0u, 0u};
        uint4* p = (uint4*)patch;
        constexpr int NQ = 8 * 16 * PST * 2 / 16;
        for (int i = tid; i < NQ; i += 256) p[i] = z;
    }
    // stage weights transposed: wt[n][k] = W[k*COUT + n], f16, zero pad k>=K
    for (int i = tid; i < COUT * KPAD; i += 256) {
        int nn = i / KPAD, k = i % KPAD;
        wt[i] = (k < K) ? (_Float16)Wg[k * COUT + nn] : (_Float16)0.f;
    }
    __syncthreads();

    // gather patches: (128 voxels x 9 taps), CIN f16 per tap = 1-2 x b128 copies
    const int wg_base = blockIdx.x * 128;
    for (int p = tid; p < 128 * 9; p += 256) {
        int vl = p / 9, tap = p % 9;
        int vox = wg_base + vl;
        if (vox < n) {
            int j = nbr[(size_t)vox * 9 + tap];
            if (j >= 0) {
                const uint4* s = (const uint4*)(xh + (size_t)j * CIN);
                uint4* d = (uint4*)(patch + vl * PST + tap * CIN);
#pragma unroll
                for (int q = 0; q < CIN / 8; ++q) d[q] = s[q];
            }
        }
    }
    __syncthreads();

    const int lane = tid & 31;
    const int wv   = tid >> 5;
    const int m    = lane & 15;   // A row / C col selector
    const int hs   = lane >> 4;   // half select
    const _Float16* prow = patch + (wv * 16 + m) * PST;
    const _Float16* wcol = wt + m * KPAD;  // column (lane&15) of current N-block

#pragma unroll
    for (int nb = 0; nb < COUT / 16; ++nb) {
        v8f acc = {};
        const _Float16* wr = wcol + nb * 16 * KPAD;
#pragma unroll
        for (int kc = 0; kc < KPAD / 32; ++kc) {
            union { uint4 q[2]; v16h v; } a, b;
            int ak = kc * 32 + hs * 8;   // A lane K-set: {ak..ak+7, ak+16..ak+23}
            a.q[0] = *(const uint4*)(prow + ak);
            a.q[1] = *(const uint4*)(prow + ak + 16);
            int bk = kc * 32 + hs * 8;   // B lane K-set: {bk..bk+7, bk+16..bk+23}
            b.q[0] = *(const uint4*)(wr + bk);
            b.q[1] = *(const uint4*)(wr + bk + 16);
            acc = __builtin_amdgcn_wmma_f32_16x16x32_f16(
                false, a.v, false, b.v, (short)0, acc, false, false);
        }
#pragma unroll
        for (int r = 0; r < 8; ++r) {
            int vox = wg_base + wv * 16 + r + hs * 8;  // C: VGPR r -> row r+8*hs
            if (vox < n) y[(size_t)vox * COUT + nb * 16 + m] = acc[r];
        }
    }
}

// ---------------------------------------------------------------- BN (deterministic 2-stage)
template <int C>
__global__ __launch_bounds__(256) void bn_stats1_kernel(const float* __restrict__ y, int n,
                                                        float* __restrict__ pS,
                                                        float* __restrict__ pQ) {
    constexpr int R = 256 / C;
    int c = threadIdx.x % C, rg = threadIdx.x / C;
    float s = 0.f, q = 0.f;
    for (long i = (long)blockIdx.x * R + rg; i < n; i += (long)gridDim.x * R) {
        float v = y[i * C + c];
        s += v; q += v * v;
    }
    __shared__ float sS[256], sQ[256];
    sS[threadIdx.x] = s; sQ[threadIdx.x] = q;
    __syncthreads();
    for (int off = 128; off >= C; off >>= 1) {  // off multiple of C -> same channel
        if (threadIdx.x < off) {
            sS[threadIdx.x] += sS[threadIdx.x + off];
            sQ[threadIdx.x] += sQ[threadIdx.x + off];
        }
        __syncthreads();
    }
    if (threadIdx.x < C) {
        pS[blockIdx.x * C + threadIdx.x] = sS[threadIdx.x];
        pQ[blockIdx.x * C + threadIdx.x] = sQ[threadIdx.x];
    }
}

template <int C>
__global__ __launch_bounds__(256) void bn_stats2_kernel(const float* __restrict__ pS,
                                                        const float* __restrict__ pQ, int G, int n,
                                                        const float* __restrict__ g,
                                                        const float* __restrict__ b,
                                                        float* __restrict__ scale,
                                                        float* __restrict__ shiftv) {
    int c = threadIdx.x;
    if (c >= C) return;
    float S = 0.f, Q = 0.f;
    for (int i = 0; i < G; ++i) { S += pS[i * C + c]; Q += pQ[i * C + c]; }
    float mean = S / (float)n;
    float var  = Q / (float)n - mean * mean;
    float sc   = g[c] * rsqrtf(var + 1e-5f);
    scale[c]  = sc;
    shiftv[c] = b[c] - mean * sc;
}

template <int C, bool POOL>
__global__ __launch_bounds__(256) void bn_apply_kernel(const float* __restrict__ y,
                                                       const float* __restrict__ scale,
                                                       const float* __restrict__ shiftv,
                                                       _Float16* __restrict__ xh,
                                                       const int* __restrict__ idx,
                                                       unsigned* __restrict__ pooled, long total) {
    for (long e = (long)blockIdx.x * 256 + threadIdx.x; e < total; e += (long)gridDim.x * 256) {
        int c = (int)(e % C);
        float v = fmaxf(y[e] * scale[c] + shiftv[c], 0.f);
        if (POOL) {
            long i = e / C;
            int bb = idx[i * 3];
            atomicMax(&pooled[bb * 32 + c], __float_as_uint(v));  // exact, order-free (v>=0)
        } else {
            xh[e] = (_Float16)v;
        }
    }
}

__global__ void init_pooled_kernel(unsigned* pooled) {
    if (threadIdx.x < 512) pooled[threadIdx.x] = 0u;  // bits of +0.0f
}

// ---------------------------------------------------------------- FC: 16x32 @ 32x128, 8 WMMAs
__global__ __launch_bounds__(256) void fc_kernel(const float* __restrict__ pooled,
                                                 const float* __restrict__ Wfc,
                                                 const float* __restrict__ bfc,
                                                 float* __restrict__ out) {
    __shared__ __align__(16) _Float16 ah[16 * 32];
    __shared__ __align__(16) _Float16 bh[128 * 32];  // [n][k]
    int tid = threadIdx.x;
    for (int i = tid; i < 512; i += 256) ah[i] = (_Float16)pooled[i];
    for (int i = tid; i < 128 * 32; i += 256) {
        int nn = i >> 5, k = i & 31;
        bh[i] = (_Float16)Wfc[k * 128 + nn];
    }
    __syncthreads();
    int lane = tid & 31, wv = tid >> 5;
    int m = lane & 15, hs = lane >> 4;
    union { uint4 q[2]; v16h v; } a, b;
    a.q[0] = *(const uint4*)(ah + m * 32 + hs * 8);
    a.q[1] = *(const uint4*)(ah + m * 32 + hs * 8 + 16);
    const _Float16* wr = bh + (wv * 16 + m) * 32 + hs * 8;
    b.q[0] = *(const uint4*)(wr);
    b.q[1] = *(const uint4*)(wr + 16);
    v8f acc = {};
    acc = __builtin_amdgcn_wmma_f32_16x16x32_f16(false, a.v, false, b.v, (short)0, acc,
                                                 false, false);
    float bias = bfc[wv * 16 + m];
#pragma unroll
    for (int r = 0; r < 8; ++r) {
        int row = r + hs * 8;
        out[row * 128 + wv * 16 + m] = fmaxf(acc[r] + bias, 0.f);
    }
}

// ---------------------------------------------------------------- launch
extern "C" void kernel_launch(void* const* d_in, const int* in_sizes, int n_in,
                              void* d_out, int out_size, void* d_ws, size_t ws_size,
                              hipStream_t stream) {
    const float* feats = (const float*)d_in[0];
    const int*   idx   = (const int*)d_in[1];
    const float* W1    = (const float*)d_in[2];
    const float* g1    = (const float*)d_in[3];
    const float* b1    = (const float*)d_in[4];
    const float* W2    = (const float*)d_in[5];
    const float* g2    = (const float*)d_in[6];
    const float* b2    = (const float*)d_in[7];
    const float* W3    = (const float*)d_in[8];
    const float* g3    = (const float*)d_in[9];
    const float* b3    = (const float*)d_in[10];
    const float* Wfc   = (const float*)d_in[11];
    const float* bfc   = (const float*)d_in[12];
    float* out = (float*)d_out;

    const int N = in_sizes[0];  // feats is [N,1]
    const int MAP_TOTAL = GRID_B * GRID_H * GRID_W;
    const int G = 512;  // stats stage-1 blocks

    // workspace carve-out
    char* ws = (char*)d_ws;
    size_t off = 0;
    auto alloc = [&](size_t bytes) {
        void* p = ws + off;
        off = (off + bytes + 255) & ~(size_t)255;
        return p;
    };
    int*      map    = (int*)alloc(sizeof(int) * (size_t)MAP_TOTAL);
    int*      nbr    = (int*)alloc(sizeof(int) * (size_t)N * 9);
    float*    ybuf   = (float*)alloc(sizeof(float) * (size_t)N * 32);
    _Float16* x1h    = (_Float16*)alloc(sizeof(_Float16) * (size_t)N * 8);
    _Float16* x2h    = (_Float16*)alloc(sizeof(_Float16) * (size_t)N * 16);
    float*    pS     = (float*)alloc(sizeof(float) * (size_t)G * 32);
    float*    pQ     = (float*)alloc(sizeof(float) * (size_t)G * 32);
    float*    scale  = (float*)alloc(sizeof(float) * 32);
    float*    shiftv = (float*)alloc(sizeof(float) * 32);
    unsigned* pooled = (unsigned*)alloc(sizeof(unsigned) * 512);
    (void)ws_size; (void)n_in; (void)out_size;

    const int gN   = (N + 255) / 256;
    const int gT   = (N + 127) / 128;  // 16-voxel tiles, 8 per workgroup

    // coordinate map + shared neighbor table (identical for all subm layers)
    init_map_kernel<<<(MAP_TOTAL + 255) / 256, 256, 0, stream>>>(map, MAP_TOTAL);
    scatter_map_kernel<<<gN, 256, 0, stream>>>(idx, map, N);
    build_nbr_kernel<<<gN, 256, 0, stream>>>(idx, map, nbr, N);

    // layer 1: scalar conv (cin=1), BN+ReLU -> f16
    conv1_kernel<<<gN, 256, 0, stream>>>(feats, nbr, W1, ybuf, N);
    bn_stats1_kernel<8><<<G, 256, 0, stream>>>(ybuf, N, pS, pQ);
    bn_stats2_kernel<8><<<1, 256, 0, stream>>>(pS, pQ, G, N, g1, b1, scale, shiftv);
    bn_apply_kernel<8, false><<<2048, 256, 0, stream>>>(ybuf, scale, shiftv, x1h, idx,
                                                        pooled, (long)N * 8);

    // layer 2: WMMA implicit GEMM (K=72 -> 96), BN+ReLU -> f16
    conv_wmma_kernel<8, 16, 96><<<gT, 256, 0, stream>>>(x1h, nbr, W2, ybuf, N);
    bn_stats1_kernel<16><<<G, 256, 0, stream>>>(ybuf, N, pS, pQ);
    bn_stats2_kernel<16><<<1, 256, 0, stream>>>(pS, pQ, G, N, g2, b2, scale, shiftv);
    bn_apply_kernel<16, false><<<2048, 256, 0, stream>>>(ybuf, scale, shiftv, x2h, idx,
                                                         pooled, (long)N * 16);

    // layer 3: WMMA implicit GEMM (K=144 -> 160), BN+ReLU fused with segment-max pool
    conv_wmma_kernel<16, 32, 160><<<gT, 256, 0, stream>>>(x2h, nbr, W3, ybuf, N);
    bn_stats1_kernel<32><<<G, 256, 0, stream>>>(ybuf, N, pS, pQ);
    bn_stats2_kernel<32><<<1, 256, 0, stream>>>(pS, pQ, G, N, g3, b3, scale, shiftv);
    init_pooled_kernel<<<1, 512, 0, stream>>>(pooled);
    bn_apply_kernel<32, true><<<2048, 256, 0, stream>>>(ybuf, scale, shiftv, nullptr, idx,
                                                        pooled, (long)N * 32);

    // FC head: one workgroup, 8 WMMAs
    fc_kernel<<<1, 256, 0, stream>>>((const float*)pooled, Wfc, bfc, out);
}
